// RetrievalManifold_67740224193300
// MI455X (gfx1250) — compile-verified, matrix operand order checked by reading, not run on hardware
//
#include <hip/hip_runtime.h>

typedef __attribute__((ext_vector_type(16))) _Float16 v16h;
typedef __attribute__((ext_vector_type(8)))  float    v8f;
typedef __attribute__((ext_vector_type(4)))  float    v4f;

#define EPSF 1e-5f

__device__ __forceinline__ float sigm(float x){ return 1.0f/(1.0f+__expf(-x)); }
__device__ __forceinline__ float softplusf(float x){ return x > 20.0f ? x : log1pf(__expf(x)); }

// sum across the 16 lanes of a half-wave (lanes 0-15 or 16-31); wave32 shuffle,
// xor masks 8..1 never cross the 16-lane boundary.
__device__ __forceinline__ float halfsum(float v){
#pragma unroll
  for (int m = 8; m >= 1; m >>= 1) v += __shfl_xor(v, m, 32);
  return v;
}

// Pack W (Ntot x Ktot, row-major fp32) into f16 WMMA B-operand lane layout:
// dst[((t*ksteps + k)*32 + lane)*16 + i], lane%16 = N column within the tile,
// halves follow the CDNA5 16-bit operand K striping.
__global__ void pack_w_kernel(const float* __restrict__ W, _Float16* __restrict__ dst,
                              int Ntot, int Ktot, int ntiles, int ksteps){
  int idx = blockIdx.x*blockDim.x + threadIdx.x;
  int total = ntiles*ksteps*512;
  if (idx >= total) return;
  int i    = idx & 15;
  int lane = (idx >> 4) & 31;
  int rest = idx >> 9;
  int k = rest % ksteps;
  int t = rest / ksteps;
  int n  = t*16 + (lane & 15);
  int kb = (lane >> 4) << 3;
  int kk = 32*k + ((i < 8) ? (i + kb) : (i + 8 + kb));
  float v = (n < Ntot && kk < Ktot) ? W[(size_t)n*Ktot + kk] : 0.0f;
  dst[idx] = (_Float16)v;
}

// Gather a 16x32 f16 A-operand tile from an fp32 LDS row buffer.
__device__ __forceinline__ v16h loadA16(const float* rowptr, int kb){
  v16h a;
#pragma unroll
  for (int i = 0; i < 8; ++i) a[i]   = (_Float16)rowptr[kb + i];
#pragma unroll
  for (int i = 0; i < 8; ++i) a[8+i] = (_Float16)rowptr[kb + 16 + i];
  return a;
}

__device__ __forceinline__ v8f zero8(){
  v8f z;
#pragma unroll
  for (int i = 0; i < 8; ++i) z[i] = 0.0f;
  return z;
}

__global__ __launch_bounds__(256)
void fused_mamba_kernel(const float* __restrict__ x,
                        const _Float16* __restrict__ pw_in,
                        const float* __restrict__ b_in,
                        const float* __restrict__ ln_in_g, const float* __restrict__ ln_in_b,
                        const _Float16* __restrict__ pinw0, const _Float16* __restrict__ pinw1,
                        const float* __restrict__ Dp0, const float* __restrict__ Dp1,
                        const _Float16* __restrict__ poutw0, const _Float16* __restrict__ poutw1,
                        const float* __restrict__ ln_g0, const float* __restrict__ ln_b0,
                        const float* __restrict__ ln_g1, const float* __restrict__ ln_b1,
                        const float* __restrict__ fn_g, const float* __restrict__ fn_b,
                        const _Float16* __restrict__ pshw1, const float* __restrict__ sh_b1,
                        const _Float16* __restrict__ pshw2, const float* __restrict__ sh_b2,
                        const float* __restrict__ k_w, const float* __restrict__ k_b,
                        const float* __restrict__ ki_w, const float* __restrict__ ki_b,
                        float* __restrict__ out)
{
  const int tid  = threadIdx.x;
  const int lane = tid & 31;
  const int wave = tid >> 5;
  const int row0 = blockIdx.x * 16;

  const int arow = lane & 15;          // A/C row within half-wave
  const int kb   = (lane >> 4) << 3;   // K sub-group offset
  const int rb   = (lane >> 4) << 3;   // C/D row base

  // reduction geometry: each half-wave owns one row, each lane owns 8 columns
  const int lrow = wave*2 + (lane >> 4);   // 0..15
  const int c0   = (lane & 15) * 8;        // 0..120

  __shared__ float hbuf[16][128];
  __shared__ float projbuf[16][592];   // 577 padded to 37*16
  __shared__ float gbuf[16][256];
  __shared__ float xsdt[16][256];      // silu(x_ssm)*dt
  __shared__ float outbuf[16][128];
  __shared__ float g1buf[16][64];
  __shared__ float xbuf[16][64];       // x padded 58 -> 64
  __shared__ float rowdt[16];
  __shared__ float rowcoef[16];

  // ---- load x tile (zero-pad K 58->64) ----
  for (int e = tid; e < 16*64; e += 256){
    int r = e >> 6, c = e & 63;
    xbuf[r][c] = (c < 58) ? x[(size_t)(row0 + r)*58 + c] : 0.0f;
  }
  __syncthreads();

  // ---- h0 = x @ w_in.T + b_in : N=128 -> 8 tiles (one per wave); K=64 -> 2 steps
  {
    v8f acc = zero8();
#pragma unroll
    for (int k = 0; k < 2; ++k){
      v16h a = loadA16(&xbuf[arow][k*32], kb);
      v16h b = *(const v16h*)(pw_in + ((size_t)(wave*2 + k)*32 + lane)*16);
      acc = __builtin_amdgcn_wmma_f32_16x16x32_f16(false, a, false, b, (short)0, acc, false, false);
    }
    int col = wave*16 + (lane & 15);
#pragma unroll
    for (int i = 0; i < 8; ++i) hbuf[rb + i][col] = acc[i] + b_in[col];
  }
  __syncthreads();

  // ---- LN(h0): all 8 waves, half-wave per row, shuffle reduction ----
  {
    float vals[8];
#pragma unroll
    for (int j = 0; j < 8; ++j) vals[j] = hbuf[lrow][c0 + j];
    float s = 0.f;
#pragma unroll
    for (int j = 0; j < 8; ++j) s += vals[j];
    float m = halfsum(s) * (1.0f/128.0f);
    float vv = 0.f;
#pragma unroll
    for (int j = 0; j < 8; ++j){ float d = vals[j] - m; vv += d*d; }
    float inv = rsqrtf(halfsum(vv)*(1.0f/128.0f) + EPSF);
#pragma unroll
    for (int j = 0; j < 8; ++j)
      hbuf[lrow][c0 + j] = (vals[j] - m)*inv*ln_in_g[c0 + j] + ln_in_b[c0 + j];
  }
  __syncthreads();

  const _Float16* pinw[2]  = {pinw0, pinw1};
  const _Float16* poutw[2] = {poutw0, poutw1};
  const float* Dp[2]  = {Dp0, Dp1};
  const float* lng[2] = {ln_g0, ln_g1};
  const float* lnb[2] = {ln_b0, ln_b1};
  const size_t stOff[2] = {90112ull, 67198976ull}; // states0 / states1 float offsets

  for (int li = 0; li < 2; ++li){
    // ---- proj = h @ in_w.T : 37 N-tiles over 8 waves; K=128 -> 4 steps ----
    for (int t = wave; t < 37; t += 8){
      v8f acc = zero8();
#pragma unroll
      for (int k = 0; k < 4; ++k){
        v16h a = loadA16(&hbuf[arow][k*32], kb);
        v16h b = *(const v16h*)(pinw[li] + ((size_t)(t*4 + k)*32 + lane)*16);
        acc = __builtin_amdgcn_wmma_f32_16x16x32_f16(false, a, false, b, (short)0, acc, false, false);
      }
      int col = t*16 + (lane & 15);
#pragma unroll
      for (int i = 0; i < 8; ++i) projbuf[rb + i][col] = acc[i];
    }
    __syncthreads();

    // ---- per-row scalars: dt = softplus(proj[576]); S = B.C (parallel) ----
    {
      int sidx = lane & 15;
      float p = projbuf[lrow][512 + sidx]*projbuf[lrow][544 + sidx]
              + projbuf[lrow][528 + sidx]*projbuf[lrow][560 + sidx];
      float S = halfsum(p);
      if ((lane & 15) == 0){
        float dt = softplusf(projbuf[lrow][576]);
        rowdt[lrow]   = dt;
        rowcoef[lrow] = dt * S;
      }
    }
    __syncthreads();

    // ---- elementwise: x_ssm = silu(proj[256+d]); g = x_ssm*(dt*S + Dp)*silu(z) ----
    for (int r = 0; r < 16; ++r){
      int d = tid;                      // 256 threads == 256 channels
      float xr = projbuf[r][256 + d];
      float xs = xr * sigm(xr);
      xsdt[r][d] = xs * rowdt[r];
      float z  = projbuf[r][d];
      gbuf[r][d] = xs * (rowcoef[r] + Dp[li][d]) * (z * sigm(z));
    }
    __syncthreads();

    // ---- state write: 16x256x32 fp32 contiguous, coalesced NT b128 stores ----
    {
      float* sb = out + stOff[li] + (size_t)row0 * (256*32);
#pragma unroll 4
      for (int it = 0; it < 128; ++it){
        int q  = it*256 + tid;          // float4 index, lanes -> consecutive 16B
        int s4 = q & 7;
        int d  = (q >> 3) & 255;
        int r  = q >> 11;
        float xv = xsdt[r][d];
        v4f bv = *(const v4f*)&projbuf[r][512 + s4*4];
        v4f o  = bv * xv;
        __builtin_nontemporal_store(o, (v4f*)(sb + (size_t)q*4));
      }
    }

    // ---- outp = g @ out_w.T : N=128 -> 8 tiles (one per wave); K=256 -> 8 steps ----
    {
      v8f acc = zero8();
#pragma unroll
      for (int k = 0; k < 8; ++k){
        v16h a = loadA16(&gbuf[arow][k*32], kb);
        v16h b = *(const v16h*)(poutw[li] + ((size_t)(wave*8 + k)*32 + lane)*16);
        acc = __builtin_amdgcn_wmma_f32_16x16x32_f16(false, a, false, b, (short)0, acc, false, false);
      }
      int col = wave*16 + (lane & 15);
#pragma unroll
      for (int i = 0; i < 8; ++i) outbuf[rb + i][col] = acc[i];
    }
    __syncthreads();

    // ---- h = LN(h + outp): parallel ----
    {
      float vals[8];
#pragma unroll
      for (int j = 0; j < 8; ++j) vals[j] = hbuf[lrow][c0 + j] + outbuf[lrow][c0 + j];
      float s = 0.f;
#pragma unroll
      for (int j = 0; j < 8; ++j) s += vals[j];
      float m = halfsum(s) * (1.0f/128.0f);
      float vv = 0.f;
#pragma unroll
      for (int j = 0; j < 8; ++j){ float d = vals[j] - m; vv += d*d; }
      float inv = rsqrtf(halfsum(vv)*(1.0f/128.0f) + EPSF);
      __syncthreads();   // all reads of hbuf done before overwrite
#pragma unroll
      for (int j = 0; j < 8; ++j)
        hbuf[lrow][c0 + j] = (vals[j] - m)*inv*lng[li][c0 + j] + lnb[li][c0 + j];
    }
    __syncthreads();
  }

  // ---- final LN (fn): parallel ----
  {
    float vals[8];
#pragma unroll
    for (int j = 0; j < 8; ++j) vals[j] = hbuf[lrow][c0 + j];
    float s = 0.f;
#pragma unroll
    for (int j = 0; j < 8; ++j) s += vals[j];
    float m = halfsum(s) * (1.0f/128.0f);
    float vv = 0.f;
#pragma unroll
    for (int j = 0; j < 8; ++j){ float d = vals[j] - m; vv += d*d; }
    float inv = rsqrtf(halfsum(vv)*(1.0f/128.0f) + EPSF);
#pragma unroll
    for (int j = 0; j < 8; ++j)
      hbuf[lrow][c0 + j] = (vals[j] - m)*inv*fn_g[c0 + j] + fn_b[c0 + j];
  }
  __syncthreads();

  // ---- head GEMM1: g1 = gelu(h @ sh_w1.T + b1), N=64 -> 4 tiles (waves 0..3); K=128
  if (wave < 4){
    v8f acc = zero8();
#pragma unroll
    for (int k = 0; k < 4; ++k){
      v16h a = loadA16(&hbuf[arow][k*32], kb);
      v16h b = *(const v16h*)(pshw1 + ((size_t)(wave*4 + k)*32 + lane)*16);
      acc = __builtin_amdgcn_wmma_f32_16x16x32_f16(false, a, false, b, (short)0, acc, false, false);
    }
    int col = wave*16 + (lane & 15);
#pragma unroll
    for (int i = 0; i < 8; ++i){
      float u = acc[i] + sh_b1[col];
      g1buf[rb + i][col] = 0.5f*u*(1.0f + erff(u*0.70710678118654752f)); // exact gelu
    }
  }

  // ---- K / K_internal heads: all waves, half-wave per row, shuffle reduction ----
  {
    float a0 = 0.f, a1 = 0.f;
#pragma unroll
    for (int j = 0; j < 8; ++j){
      float hv = hbuf[lrow][c0 + j];
      a0 += hv * k_w[c0 + j];
      a1 += hv * ki_w[c0 + j];
    }
    a0 = halfsum(a0);
    a1 = halfsum(a1);
    if ((lane & 15) == 0){
      out[73728 + row0 + lrow] = sigm(a0 + k_b[0]);   // K
      out[81920 + row0 + lrow] = sigm(a1 + ki_b[0]);  // K_internal
    }
  }
  __syncthreads();

  // ---- head GEMM2: scaling = softplus(g1 @ sh_w2.T + b2), N=9 (one padded tile); K=64
  if (wave == 0){
    v8f acc = zero8();
#pragma unroll
    for (int k = 0; k < 2; ++k){
      v16h a = loadA16(&g1buf[arow][k*32], kb);
      v16h b = *(const v16h*)(pshw2 + ((size_t)k*32 + lane)*16);
      acc = __builtin_amdgcn_wmma_f32_16x16x32_f16(false, a, false, b, (short)0, acc, false, false);
    }
    int col = lane & 15;
    if (col < 9){
#pragma unroll
      for (int i = 0; i < 8; ++i){
        float u = softplusf(acc[i] + sh_b2[col]);
        out[(size_t)(row0 + rb + i)*9 + col] = u;
      }
    }
  }
}

extern "C" void kernel_launch(void* const* d_in, const int* in_sizes, int n_in,
                              void* d_out, int out_size, void* d_ws, size_t ws_size,
                              hipStream_t stream) {
  const float* x      = (const float*)d_in[0];
  const float* w_in   = (const float*)d_in[1];
  const float* b_in   = (const float*)d_in[2];
  const float* lnig   = (const float*)d_in[3];
  const float* lnib   = (const float*)d_in[4];
  const float* in_w0  = (const float*)d_in[5];
  const float* Dp0    = (const float*)d_in[7];
  const float* out_w0 = (const float*)d_in[8];
  const float* ln_g0  = (const float*)d_in[9];
  const float* ln_b0  = (const float*)d_in[10];
  const float* in_w1  = (const float*)d_in[11];
  const float* Dp1    = (const float*)d_in[13];
  const float* out_w1 = (const float*)d_in[14];
  const float* ln_g1  = (const float*)d_in[15];
  const float* ln_b1  = (const float*)d_in[16];
  const float* fn_g   = (const float*)d_in[17];
  const float* fn_b   = (const float*)d_in[18];
  const float* sh_w1  = (const float*)d_in[19];
  const float* sh_b1  = (const float*)d_in[20];
  const float* sh_w2  = (const float*)d_in[21];
  const float* sh_b2  = (const float*)d_in[22];
  const float* k_w    = (const float*)d_in[23];
  const float* k_b    = (const float*)d_in[24];
  const float* ki_w   = (const float*)d_in[25];
  const float* ki_b   = (const float*)d_in[26];
  float* out = (float*)d_out;

  // packed f16 weight arenas in workspace (offsets in halves)
  _Float16* ws     = (_Float16*)d_ws;
  _Float16* pw_in  = ws + 0;        // 8 tiles * 2 ksteps  = 8192
  _Float16* pinw0  = ws + 8192;     // 37 * 4              = 75776
  _Float16* pinw1  = ws + 83968;    // 75776
  _Float16* poutw0 = ws + 159744;   // 8 * 8               = 32768
  _Float16* poutw1 = ws + 192512;   // 32768
  _Float16* pshw1  = ws + 225280;   // 4 * 4               = 8192
  _Float16* pshw2  = ws + 233472;   // 1 * 2               = 1024

  auto pack = [&](const float* W, _Float16* dst, int Ntot, int Ktot, int ntiles, int ksteps){
    int total = ntiles*ksteps*512;
    pack_w_kernel<<<(total + 255)/256, 256, 0, stream>>>(W, dst, Ntot, Ktot, ntiles, ksteps);
  };
  pack(w_in,   pw_in,  128,  58,  8, 2);
  pack(in_w0,  pinw0,  577, 128, 37, 4);
  pack(in_w1,  pinw1,  577, 128, 37, 4);
  pack(out_w0, poutw0, 128, 256,  8, 8);
  pack(out_w1, poutw1, 128, 256,  8, 8);
  pack(sh_w1,  pshw1,   64, 128,  4, 4);
  pack(sh_w2,  pshw2,    9,  64,  1, 2);

  fused_mamba_kernel<<<512, 256, 0, stream>>>(
      x, pw_in, b_in, lnig, lnib,
      pinw0, pinw1, Dp0, Dp1, poutw0, poutw1,
      ln_g0, ln_b0, ln_g1, ln_b1, fn_g, fn_b,
      pshw1, sh_b1, pshw2, sh_b2, k_w, k_b, ki_w, ki_b,
      out);
}
